// FullConstitutiveRouter_25202868093630
// MI455X (gfx1250) — matrix operand order repaired and checked by reading, not compiled, not run on hardware
//
#include <hip/hip_runtime.h>

#define D    768
#define P    576
#define GDIM 24
#define KSEL 8
#define FPN  64
#define NA   16
#define NB   32
#define ROWS (NB * P)          // 18432
#define TEMP_INV 10.0f
#define NEGV -1000000000.0f

typedef __attribute__((ext_vector_type(16))) __bf16    v16bf;
typedef __attribute__((ext_vector_type(8)))  float     v8f;
typedef __attribute__((ext_vector_type(8)))  unsigned  v8u;   // 32B = one lane's bf16 fragment

// ---------- bf16 helpers ----------
// exact RNE (used in memory-bound scatter writers)
__device__ __forceinline__ unsigned short f2bf(float f) {
  unsigned u = __builtin_bit_cast(unsigned, f);
  unsigned r = u + 0x7FFFu + ((u >> 16) & 1u);
  return (unsigned short)(r >> 16);
}
// fast packed pair: round-to-nearest (+0x8000) then one v_perm_b32 to grab both high halves
__device__ __forceinline__ unsigned pack_bf16x2(float lo, float hi) {
  unsigned u0 = __builtin_bit_cast(unsigned, lo) + 0x8000u;
  unsigned u1 = __builtin_bit_cast(unsigned, hi) + 0x8000u;
  return __builtin_amdgcn_perm(u1, u0, 0x07060302u);  // {u1[31:16], u0[31:16]}
}
__device__ __forceinline__ v8f vzero8() {
  v8f z = {0.f, 0.f, 0.f, 0.f, 0.f, 0.f, 0.f, 0.f};
  return z;
}
__device__ __forceinline__ v16bf frag(v8u raw) { return __builtin_bit_cast(v16bf, raw); }

// B-operand swizzle: element (k, n) -> fragment-major index.
// Frag layout (16-bit B 32x16, contiguous k order): lane = 16*k[4] + (n&15), elem i = k&15.
__device__ __forceinline__ size_t bswz(int k, int n) {
  return ((size_t)((n >> 4) * (D / 32) + (k >> 5)) * 32 + ((k & 16) + (n & 15))) * 16 + (k & 15);
}

// =====================================================================
// Fold fingerprint modulation:  E[k,n] = W[n,k] + 0.1*(Wm@f + bm)[k,n]  (swizzled bf16)
// =====================================================================
__global__ void prep_e_kernel(const float* __restrict__ Wbase, const float* __restrict__ Wm,
                              const float* __restrict__ bm, const float* __restrict__ fp,
                              unsigned short* __restrict__ E) {
  int idx = blockIdx.x * blockDim.x + threadIdx.x;
  if (idx >= D * D) return;
  int k = idx / D, n = idx - k * D;
  float s = bm[idx];
  const float* r = Wm + (size_t)idx * FPN;
#pragma unroll 8
  for (int c = 0; c < FPN; ++c) s += r[c] * fp[c];
  E[bswz(k, n)] = f2bf(Wbase[(size_t)n * D + k] + 0.1f * s);
}

__global__ void prep_wot_kernel(const float* __restrict__ Wo, unsigned short* __restrict__ E) {
  int idx = blockIdx.x * blockDim.x + threadIdx.x;
  if (idx >= D * D) return;
  int k = idx / D, n = idx - k * D;
  E[bswz(k, n)] = f2bf(Wo[(size_t)n * D + k]);
}

// =====================================================================
// Scalars + cantor cnorm + fp_bias + gate (one small block, 576 threads)
// =====================================================================
__global__ void prep_misc_kernel(const float* __restrict__ fp, const float* __restrict__ Wfo,
                                 const float* __restrict__ bfo, const float* __restrict__ Wab,
                                 const float* __restrict__ bab, const float* __restrict__ aw,
                                 float* __restrict__ cn, float* __restrict__ gate,
                                 float* __restrict__ fpb) {
  __shared__ float s_scale, s_shift, s_den;
  __shared__ float sc[P];
  int t = threadIdx.x;
  if (t == 0) {
    float p0 = bfo[0], p1 = bfo[1];
    for (int c = 0; c < FPN; ++c) { p0 += Wfo[c] * fp[c]; p1 += Wfo[FPN + c] * fp[c]; }
    s_scale = (1.0f / (1.0f + __expf(-p0))) * 2.0f + 0.5f;
    s_shift = tanhf(p1) * (float)GDIM;
    gate[0] = 1.0f / (1.0f + __expf(-aw[0]));
  }
  if (t < NA) {
    float s = bab[t];
    for (int c = 0; c < FPN; ++c) s += Wab[t * FPN + c] * fp[c];
    fpb[t] = s;
  }
  __syncthreads();
  if (t < P) {
    int xg = t % GDIM, yg = t / GDIM;
    int sg = xg + yg;
    sc[t] = (float)((sg * (sg + 1)) / 2 + yg) * s_scale + s_shift;
  }
  __syncthreads();
  if (t == 0) {
    float mx = sc[0];
    for (int i = 1; i < P; ++i) mx = fmaxf(mx, sc[i]);
    s_den = fmaxf(mx, 1.0f);
  }
  __syncthreads();
  if (t < P) cn[t] = sc[t] / s_den;
}

// =====================================================================
// L2-normalize anchor embeddings
// =====================================================================
__global__ void anchor_norm_kernel(const float* __restrict__ ae, float* __restrict__ an) {
  __shared__ float red[256];
  int a = blockIdx.x, t = threadIdx.x;
  const float* r = ae + (size_t)a * D;
  float v0 = r[t], v1 = r[t + 256], v2 = r[t + 512];
  red[t] = v0 * v0 + v1 * v1 + v2 * v2;
  __syncthreads();
  for (int s = 128; s; s >>= 1) { if (t < s) red[t] += red[t + s]; __syncthreads(); }
  float inv = 1.0f / fmaxf(sqrtf(red[0]), 1e-12f);
  float* o = an + (size_t)a * D;
  o[t] = v0 * inv; o[t + 256] = v1 * inv; o[t + 512] = v2 * inv;
}

// =====================================================================
// Stage one 16x32 fp32 tile -> bf16 A-fragment layout in LDS (256 threads, 2 elems each)
// A frag layout: lane = m + 16*k[4], elem i = k&15
// Conversion: 2x v_add + 1x v_perm_b32 + 1x ds_store_b32 per element pair.
// =====================================================================
__device__ __forceinline__ void stageA16x32(unsigned short* __restrict__ dst,
                                            const float* __restrict__ xrow0, int ks) {
  int t = threadIdx.x;
  int mr = t >> 4;
  int kk = (t & 15) << 1;
  float2 xx = *(const float2*)(xrow0 + (size_t)mr * D + ks + kk);
  int lt = mr + (((kk >> 4) & 1) << 4);
  int i  = kk & 15;
  ((unsigned*)dst)[(lt * 16 + i) >> 1] = pack_bf16x2(xx.x, xx.y);
}

// =====================================================================
// WMMA GEMM:  (18432 x 768) fp32 x swizzled-bf16 (768x768) -> fp32 + bias
// 256 thr = 8 waves; block tile 64 rows x 128 cols; 4 M-subtiles/wave share one B frag.
// Double-buffered LDS A staging (one barrier per K step).
// =====================================================================
__launch_bounds__(256)
__global__ void gemm_qkv_kernel(const float* __restrict__ x,
                                const unsigned short* __restrict__ Bsw,
                                const float* __restrict__ bias,
                                float* __restrict__ C) {
  const int wave = threadIdx.x >> 5, lane = threadIdx.x & 31;
  const int m0 = blockIdx.y * 64;
  const int n0 = blockIdx.x * 8 + wave;                 // 16-col tile index
  const int bb = m0 / P, p0 = m0 - bb * P;              // never crosses batch (64 | 576)
  const float* xbase = x + ((size_t)(bb * 577 + 1 + p0)) * D;
  const v8u* B8 = (const v8u*)Bsw;
  __shared__ v8u As8[2][4 * 32];                        // 2 x (4 subtiles x 32 lanes) = 8KB
  v8f acc0 = vzero8(), acc1 = vzero8(), acc2 = vzero8(), acc3 = vzero8();

#pragma unroll
  for (int s = 0; s < 4; ++s)
    stageA16x32((unsigned short*)As8[0] + s * 512, xbase + (size_t)s * 16 * D, 0);
  __syncthreads();

  int pb = 0;
#pragma unroll 4
  for (int ks = 0; ks < D; ks += 32) {
    if (ks + 32 < D) {
#pragma unroll
      for (int s = 0; s < 4; ++s)
        stageA16x32((unsigned short*)As8[pb ^ 1] + s * 512, xbase + (size_t)s * 16 * D, ks + 32);
      __builtin_prefetch(&B8[(size_t)(n0 * (D / 32) + ((ks + 32) >> 5)) * 32 + lane], 0, 1);
    }
    v16bf bmf = frag(B8[(size_t)(n0 * (D / 32) + (ks >> 5)) * 32 + lane]);
    acc0 = __builtin_amdgcn_wmma_f32_16x16x32_bf16(false, frag(As8[pb][lane]),       false, bmf, (short)0, acc0, false, false);
    acc1 = __builtin_amdgcn_wmma_f32_16x16x32_bf16(false, frag(As8[pb][32 + lane]),  false, bmf, (short)0, acc1, false, false);
    acc2 = __builtin_amdgcn_wmma_f32_16x16x32_bf16(false, frag(As8[pb][64 + lane]),  false, bmf, (short)0, acc2, false, false);
    acc3 = __builtin_amdgcn_wmma_f32_16x16x32_bf16(false, frag(As8[pb][96 + lane]),  false, bmf, (short)0, acc3, false, false);
    __syncthreads();
    pb ^= 1;
  }

  const int n = n0 * 16 + (lane & 15);
  const float bvl = bias[n];
  const int mb = (lane < 16) ? 0 : 8;
  v8f accs[4] = {acc0, acc1, acc2, acc3};
#pragma unroll
  for (int s = 0; s < 4; ++s)
#pragma unroll
    for (int r = 0; r < 8; ++r)
      C[(size_t)(m0 + s * 16 + mb + r) * D + n] = accs[s][r] + bvl;
}

// =====================================================================
// Row L2-normalize fp32 -> swizzled bf16 strips (one wave per row)
// amode=1: A-fragment layout (q).  amode=0: B-fragment layout (k).
// =====================================================================
__launch_bounds__(256)
__global__ void rownorm_kernel(const float* __restrict__ src, unsigned short* __restrict__ dst,
                               int amode) {
  int wave = threadIdx.x >> 5, lane = threadIdx.x & 31;
  int row = blockIdx.x * 8 + wave;
  const float* s = src + (size_t)row * D;
  float v[24];
  float ss = 0.f;
#pragma unroll
  for (int c = 0; c < 24; ++c) { v[c] = s[lane + 32 * c]; ss += v[c] * v[c]; }
#pragma unroll
  for (int off = 16; off; off >>= 1) ss += __shfl_xor(ss, off, 32);
  float inv = 1.0f / fmaxf(sqrtf(ss), 1e-12f);
  int strip = row >> 4, m = row & 15;
  unsigned short* base = dst + (size_t)strip * (16 * D);
  int off16;
  if (amode) {
    int lt = m + (((lane >> 4) & 1) << 4);
    int i  = lane & 15;
    off16 = lt * 16 + i;
  } else {
    int lt = ((lane >> 4) << 4) + m;
    int i  = lane & 15;
    off16 = lt * 16 + i;
  }
#pragma unroll
  for (int c = 0; c < 24; ++c) base[c * 512 + off16] = f2bf(v[c] * inv);
}

// =====================================================================
// pooled = mean over patches
// =====================================================================
__global__ void pooled_kernel(const float* __restrict__ x, float* __restrict__ pooled) {
  int idx = blockIdx.x * 256 + threadIdx.x;
  if (idx >= NB * D) return;
  int b = idx / D, d = idx - b * D;
  const float* base = x + ((size_t)b * 577 + 1) * D + d;
  float s = 0.f;
  for (int p = 0; p < P; ++p) s += base[(size_t)p * D];
  pooled[idx] = s * (1.0f / (float)P);
}

// =====================================================================
// feat_proj + anchor affinity softmax (one block per batch)
// =====================================================================
__global__ void featproj_kernel(const float* __restrict__ pooled, const float* __restrict__ Wfp,
                                const float* __restrict__ bfp, const float* __restrict__ an,
                                const float* __restrict__ fpb, float* __restrict__ aff) {
  __shared__ float red[256];
  __shared__ float logit[NA];
  __shared__ float pl[D];
  int b = blockIdx.x, t = threadIdx.x;
  pl[t] = pooled[b * D + t];
  pl[t + 256] = pooled[b * D + t + 256];
  pl[t + 512] = pooled[b * D + t + 512];
  __syncthreads();
  float tv[3];
#pragma unroll
  for (int c = 0; c < 3; ++c) {
    int j = t + c * 256;
    float s = bfp[j];
    const float* wr = Wfp + (size_t)j * D;
    for (int i = 0; i < D; ++i) s += pl[i] * wr[i];
    tv[c] = s;
  }
  red[t] = tv[0] * tv[0] + tv[1] * tv[1] + tv[2] * tv[2];
  __syncthreads();
  for (int s = 128; s; s >>= 1) { if (t < s) red[t] += red[t + s]; __syncthreads(); }
  float inv = 1.0f / fmaxf(sqrtf(red[0]), 1e-12f);
  __syncthreads();
  for (int a = 0; a < NA; ++a) {
    float s = 0.f;
#pragma unroll
    for (int c = 0; c < 3; ++c) { int j = t + c * 256; s += tv[c] * inv * an[(size_t)a * D + j]; }
    red[t] = s;
    __syncthreads();
    for (int s2 = 128; s2; s2 >>= 1) { if (t < s2) red[t] += red[t + s2]; __syncthreads(); }
    if (t == 0) logit[a] = red[0] + 0.3f * fpb[a];
    __syncthreads();
  }
  if (t == 0) {
    float mx = logit[0];
    for (int a = 1; a < NA; ++a) mx = fmaxf(mx, logit[a]);
    float sum = 0.f, e[NA];
    for (int a = 0; a < NA; ++a) { e[a] = __expf(logit[a] - mx); sum += e[a]; }
    for (int a = 0; a < NA; ++a) aff[b * NA + a] = e[a] / sum;
  }
}

// =====================================================================
// anchor_out[b,e] = sum_a aff[b,a] * (pooled[b,:] . anchor_W[a,e,:])
// =====================================================================
__global__ void anchor_out_kernel(const float* __restrict__ pooled, const float* __restrict__ aW,
                                  const float* __restrict__ aff, float* __restrict__ aout) {
  int idx = blockIdx.x * 256 + threadIdx.x;
  if (idx >= NB * D) return;
  int b = idx / D, e = idx - b * D;
  const float* pr = pooled + (size_t)b * D;
  float s = 0.f;
  for (int a = 0; a < NA; ++a) {
    const float* wr = aW + ((size_t)a * D + e) * D;
    float dot = 0.f;
    for (int i = 0; i < D; ++i) dot += pr[i] * wr[i];
    s += aff[b * NA + a] * dot;
  }
  aout[idx] = s;
}

// =====================================================================
// Fused router: WMMA score strip (16 x 576) + affinity -> top-8 -> softmax -> v gather
// grid = (P/16, NB); 256 threads = 8 waves. q strips in A-layout, k strips in B-layout.
// =====================================================================
__launch_bounds__(256)
__global__ void router_kernel(const v8u* __restrict__ q8,
                              const v8u* __restrict__ k8,
                              const float* __restrict__ v32,
                              const float* __restrict__ cn,
                              float* __restrict__ routed,
                              float* __restrict__ oroutes,
                              float* __restrict__ oweights) {
  __shared__ v8u qs8[(D / 32) * 32];      // 24 KB q strip (frag-major)
  __shared__ float S[16 * P];             // 36 KB score strip
  __shared__ float cns[P];
  const int tile = blockIdx.x, b = blockIdx.y;
  const int wave = threadIdx.x >> 5, lane = threadIdx.x & 31;
  const int strip = b * (P / 16) + tile;
  const v8u* qsrc = q8 + (size_t)strip * (D / 32) * 32;
  for (int e = threadIdx.x; e < (D / 32) * 32; e += 256) qs8[e] = qsrc[e];
  for (int e = threadIdx.x; e < P; e += 256) cns[e] = cn[e];
  __syncthreads();

  const int mlane = lane & 15;
  for (int t = wave; t < P / 16; t += 8) {
    v8f acc = vzero8();
    const v8u* ksrc = k8 + (size_t)(b * (P / 16) + t) * (D / 32) * 32;
#pragma unroll
    for (int ks = 0; ks < D / 32; ++ks) {
      acc = __builtin_amdgcn_wmma_f32_16x16x32_bf16(
          false, frag(qs8[ks * 32 + lane]), false, frag(ksrc[ks * 32 + lane]),
          (short)0, acc, false, false);
    }
    const int pcol = t * 16 + mlane;
    const int mb = (lane < 16) ? 0 : 8;
#pragma unroll
    for (int r = 0; r < 8; ++r) {
      int mrow = mb + r;
      int prow = tile * 16 + mrow;
      float val = (prow == pcol) ? NEGV
                                 : (acc[r] + 0.3f * (1.0f - fabsf(cns[prow] - cns[pcol])));
      S[mrow * P + pcol] = val * TEMP_INV;
    }
  }
  __syncthreads();

  volatile float* Sv = S;
  for (int rr = wave; rr < 16; rr += 8) {
    const int grow = b * P + tile * 16 + rr;
    float vals[KSEL];
    int idxs[KSEL];
#pragma unroll
    for (int it = 0; it < KSEL; ++it) {
      float bv = -3.4e38f; int bi = 0;
      for (int j = lane; j < P; j += 32) {
        float s = Sv[rr * P + j];
        if (s > bv) { bv = s; bi = j; }
      }
#pragma unroll
      for (int off = 16; off; off >>= 1) {
        float ov = __shfl_xor(bv, off, 32);
        int   oi = __shfl_xor(bi, off, 32);
        if (ov > bv || (ov == bv && oi < bi)) { bv = ov; bi = oi; }
      }
      if (lane == 0) Sv[rr * P + bi] = -3.4e38f;
      asm volatile("s_wait_dscnt 0" ::: "memory");
      vals[it] = bv; idxs[it] = bi;
    }
    float m0 = vals[0], sum = 0.f, w[KSEL];
#pragma unroll
    for (int it = 0; it < KSEL; ++it) { w[it] = __expf(vals[it] - m0); sum += w[it]; }
    float inv = 1.0f / sum;
#pragma unroll
    for (int it = 0; it < KSEL; ++it) {
      if (lane == it) {
        oroutes[(size_t)grow * KSEL + it]  = (float)idxs[it];
        oweights[(size_t)grow * KSEL + it] = w[it] * inv;
      }
    }
    float accv[24];
#pragma unroll
    for (int c = 0; c < 24; ++c) accv[c] = 0.f;
    for (int it = 0; it < KSEL; ++it) {
      const float wgt = w[it] * inv;
      const float* vr = v32 + ((size_t)b * P + idxs[it]) * D;
#pragma unroll
      for (int c = 0; c < 24; ++c) accv[c] += wgt * vr[lane + 32 * c];
    }
    float* orow = routed + (size_t)grow * D;
#pragma unroll
    for (int c = 0; c < 24; ++c) orow[lane + 32 * c] = accv[c];
  }
}

// =====================================================================
// Final WMMA GEMM: routed @ Wo^T, fused epilogue -> output tensor
// =====================================================================
__launch_bounds__(256)
__global__ void gemm_final_kernel(const float* __restrict__ Rt,
                                  const unsigned short* __restrict__ Bsw,
                                  const float* __restrict__ bo,
                                  const float* __restrict__ x,
                                  const float* __restrict__ aout,
                                  const float* __restrict__ gate,
                                  float* __restrict__ out) {
  const int wave = threadIdx.x >> 5, lane = threadIdx.x & 31;
  const int m0 = blockIdx.y * 64;
  const int n0 = blockIdx.x * 8 + wave;
  const int bb = m0 / P, p0 = m0 - bb * P;
  const float* xbase = Rt + (size_t)m0 * D;
  const v8u* B8 = (const v8u*)Bsw;
  __shared__ v8u As8[2][4 * 32];
  v8f acc0 = vzero8(), acc1 = vzero8(), acc2 = vzero8(), acc3 = vzero8();

#pragma unroll
  for (int s = 0; s < 4; ++s)
    stageA16x32((unsigned short*)As8[0] + s * 512, xbase + (size_t)s * 16 * D, 0);
  __syncthreads();

  int pb = 0;
#pragma unroll 4
  for (int ks = 0; ks < D; ks += 32) {
    if (ks + 32 < D) {
#pragma unroll
      for (int s = 0; s < 4; ++s)
        stageA16x32((unsigned short*)As8[pb ^ 1] + s * 512, xbase + (size_t)s * 16 * D, ks + 32);
    }
    v16bf bmf = frag(B8[(size_t)(n0 * (D / 32) + (ks >> 5)) * 32 + lane]);
    acc0 = __builtin_amdgcn_wmma_f32_16x16x32_bf16(false, frag(As8[pb][lane]),      false, bmf, (short)0, acc0, false, false);
    acc1 = __builtin_amdgcn_wmma_f32_16x16x32_bf16(false, frag(As8[pb][32 + lane]), false, bmf, (short)0, acc1, false, false);
    acc2 = __builtin_amdgcn_wmma_f32_16x16x32_bf16(false, frag(As8[pb][64 + lane]), false, bmf, (short)0, acc2, false, false);
    acc3 = __builtin_amdgcn_wmma_f32_16x16x32_bf16(false, frag(As8[pb][96 + lane]), false, bmf, (short)0, acc3, false, false);
    __syncthreads();
    pb ^= 1;
  }

  const int n = n0 * 16 + (lane & 15);
  const float g = gate[0];
  const float bol = bo[n];
  const float ga = g * aout[(size_t)bb * D + n];
  const int mb = (lane < 16) ? 0 : 8;
  v8f accs[4] = {acc0, acc1, acc2, acc3};
#pragma unroll
  for (int s = 0; s < 4; ++s)
#pragma unroll
    for (int r = 0; r < 8; ++r) {
      int prow = p0 + s * 16 + mb + r;
      size_t o = ((size_t)bb * 577 + 1 + prow) * D + n;
      out[o] = x[o] + accs[s][r] + bol + ga;
    }
}

__global__ void cls_copy_kernel(const float* __restrict__ x, float* __restrict__ out) {
  int idx = blockIdx.x * 256 + threadIdx.x;
  if (idx >= NB * D) return;
  int b = idx / D, d = idx - b * D;
  out[(size_t)b * 577 * D + d] = x[(size_t)b * 577 * D + d];
}

// =====================================================================
extern "C" void kernel_launch(void* const* d_in, const int* in_sizes, int n_in,
                              void* d_out, int out_size, void* d_ws, size_t ws_size,
                              hipStream_t stream) {
  const float* x   = (const float*)d_in[0];
  const float* fp  = (const float*)d_in[1];
  const float* Wq  = (const float*)d_in[2];
  const float* bq  = (const float*)d_in[3];
  const float* Wk  = (const float*)d_in[4];
  const float* bk  = (const float*)d_in[5];
  const float* Wv  = (const float*)d_in[6];
  const float* bv  = (const float*)d_in[7];
  const float* Wo  = (const float*)d_in[8];
  const float* bo  = (const float*)d_in[9];
  const float* Wqm = (const float*)d_in[10];
  const float* bqm = (const float*)d_in[11];
  const float* Wkm = (const float*)d_in[12];
  const float* bkm = (const float*)d_in[13];
  const float* Wvm = (const float*)d_in[14];
  const float* bvm = (const float*)d_in[15];
  const float* Wfo = (const float*)d_in[16];
  const float* bfo = (const float*)d_in[17];
  const float* ae  = (const float*)d_in[18];
  const float* aW  = (const float*)d_in[19];
  const float* Wab = (const float*)d_in[20];
  const float* bab = (const float*)d_in[21];
  const float* Wfp = (const float*)d_in[22];
  const float* bfp = (const float*)d_in[23];
  const float* aw  = (const float*)d_in[24];

  char* ws = (char*)d_ws;
  float* wsf = (float*)d_ws;
  float* cn     = wsf + 0;        // 576
  float* gate   = wsf + 1024;     // 1
  float* fpb    = wsf + 1056;     // 16
  float* aff    = wsf + 2048;     // 512
  float* pooled = wsf + 4096;     // 24576
  float* aout   = wsf + 32768;    // 24576
  float* anrm   = wsf + 65536;    // 12288
  unsigned short* Eq  = (unsigned short*)(ws + ((size_t)1 << 20));
  unsigned short* Ek  = (unsigned short*)(ws + ((size_t)3 << 20));
  unsigned short* Ev  = (unsigned short*)(ws + ((size_t)5 << 20));
  unsigned short* WoT = (unsigned short*)(ws + ((size_t)7 << 20));
  unsigned short* qbf = (unsigned short*)(ws + ((size_t)9 << 20));
  unsigned short* kbf = (unsigned short*)(ws + ((size_t)38 << 20));
  float* q32 = (float*)(ws + ((size_t)67 << 20));
  float* k32 = (float*)(ws + ((size_t)124 << 20));
  float* v32 = (float*)(ws + ((size_t)181 << 20));
  float* routed = q32;  // q32 dead after rownorm -> reuse

  float* oroutes  = (float*)d_out;
  float* oweights = oroutes + (size_t)NB * P * KSEL;
  float* output   = oweights + (size_t)NB * P * KSEL;

  dim3 b256(256);
  int gE = (D * D + 255) / 256;
  prep_e_kernel<<<gE, b256, 0, stream>>>(Wq, Wqm, bqm, fp, Eq);
  prep_e_kernel<<<gE, b256, 0, stream>>>(Wk, Wkm, bkm, fp, Ek);
  prep_e_kernel<<<gE, b256, 0, stream>>>(Wv, Wvm, bvm, fp, Ev);
  prep_wot_kernel<<<gE, b256, 0, stream>>>(Wo, WoT);
  prep_misc_kernel<<<1, 576, 0, stream>>>(fp, Wfo, bfo, Wab, bab, aw, cn, gate, fpb);
  anchor_norm_kernel<<<NA, b256, 0, stream>>>(ae, anrm);

  dim3 gGemm(D / 128, ROWS / 64);
  gemm_qkv_kernel<<<gGemm, b256, 0, stream>>>(x, Eq, bq, q32);
  gemm_qkv_kernel<<<gGemm, b256, 0, stream>>>(x, Ek, bk, k32);
  gemm_qkv_kernel<<<gGemm, b256, 0, stream>>>(x, Ev, bv, v32);

  rownorm_kernel<<<ROWS / 8, b256, 0, stream>>>(q32, qbf, 1);
  rownorm_kernel<<<ROWS / 8, b256, 0, stream>>>(k32, kbf, 0);

  pooled_kernel<<<(NB * D) / 256, b256, 0, stream>>>(x, pooled);
  featproj_kernel<<<NB, b256, 0, stream>>>(pooled, Wfp, bfp, anrm, fpb, aff);
  anchor_out_kernel<<<(NB * D) / 256, b256, 0, stream>>>(pooled, aW, aff, aout);

  dim3 gRoute(P / 16, NB);
  router_kernel<<<gRoute, b256, 0, stream>>>((const v8u*)qbf, (const v8u*)kbf, v32, cn,
                                             routed, oroutes, oweights);

  gemm_final_kernel<<<gGemm, b256, 0, stream>>>(routed, WoT, bo, x, aout, gate, output);
  cls_copy_kernel<<<(NB * D) / 256, b256, 0, stream>>>(x, output);
}
